// PointNetEncoder_72713796322156
// MI455X (gfx1250) — matrix operand order repaired and checked by reading, not compiled, not run on hardware
//
#include <hip/hip_runtime.h>
#include <hip/hip_bf16.h>

typedef _Float16 half_t;
typedef __attribute__((ext_vector_type(8)))  _Float16 v8h;
typedef __attribute__((ext_vector_type(16))) _Float16 v16h;
typedef __attribute__((ext_vector_type(8)))  float    v8f;

#define BN 64
#define P1 512
#define S1 103
#define S2 26
#define KN 64
#define EPSF 1e-5f
#define BIGF 3.0e38f

// ---------------------------------------------------------------- weight f32 -> f16, pre-swizzled
// into WMMA B-fragment order. Fragment (ct, ks): 32 lanes x 16 halfs contiguous.
// Element e of lane l  <->  W[ks*32 + (l>>4)*8 + (e&7) + (e>>3)*16][ct*16 + (l&15)]  (zero if k >= kin).
// Fragment linear index f = ct * nKs + ks  (K-steps contiguous for streaming).
__global__ void pack_w_kernel(const float* __restrict__ src,
                              half_t* __restrict__ dst,
                              int kin, int cout, int nKs) {
  int i = blockIdx.x * blockDim.x + threadIdx.x;
  if (i >= nKs * 32 * (cout >> 4) * 16) return;
  int e    = i & 15;
  int lane = (i >> 4) & 31;
  int f    = i >> 9;
  int ks   = f % nKs;
  int ct   = f / nKs;
  int k = ks * 32 + (lane >> 4) * 8 + (e & 7) + (e >> 3) * 16;
  int c = ct * 16 + (lane & 15);
  dst[i] = (k < kin) ? (half_t)src[(size_t)k * cout + c] : (half_t)0.0f;
}

// ---------------------------------------------------------------- farthest point sampling
__global__ void fps_kernel(const float* __restrict__ pos,
                           float* __restrict__ centers,
                           int Pn, int Sn) {
  __shared__ float redv[512];
  __shared__ int   redi[512];
  __shared__ float lastp[3];
  __shared__ int   s_last;
  int b = blockIdx.x, t = threadIdx.x;
  bool act = t < Pn;
  float px = 0.f, py = 0.f, pz = 0.f;
  if (act) {
    const float* p = pos + ((size_t)b * Pn + t) * 3;
    px = p[0]; py = p[1]; pz = p[2];
  }
  float dmin = BIGF;   // acts as +inf
  int last = 0;
  for (int s = 0; s < Sn; ++s) {
    if (t < 3) {
      float v = pos[((size_t)b * Pn + last) * 3 + t];
      lastp[t] = v;
      centers[((size_t)b * Sn + s) * 3 + t] = v;
    }
    __syncthreads();
    if (act) {
      float dx = px - lastp[0], dy = py - lastp[1], dz = pz - lastp[2];
      dmin = fminf(dmin, dx * dx + dy * dy + dz * dz);
    }
    redv[t] = act ? dmin : -BIGF;
    redi[t] = t;
    __syncthreads();
    for (int off = 256; off > 0; off >>= 1) {
      if (t < off) {
        float ov = redv[t + off]; int oi = redi[t + off];
        if (ov > redv[t] || (ov == redv[t] && oi < redi[t])) { redv[t] = ov; redi[t] = oi; }
      }
      __syncthreads();
    }
    if (t == 0) s_last = redi[0];
    __syncthreads();
    last = s_last;
  }
}

// ---------------------------------------------------------------- radius-limited 64-NN
__global__ void knn_kernel(const float* __restrict__ pos,
                           const float* __restrict__ centers,
                           int* __restrict__ nbr,
                           int Pn, int Sn, float r2) {
  __shared__ float d2s[512];
  __shared__ float rv[128];
  __shared__ int   ri[128];
  int cidx = blockIdx.x;
  int b = cidx / Sn;
  int t = threadIdx.x;
  float cx = centers[(size_t)cidx * 3 + 0];
  float cy = centers[(size_t)cidx * 3 + 1];
  float cz = centers[(size_t)cidx * 3 + 2];
  for (int p = t; p < 512; p += 128) {
    float d = BIGF;
    if (p < Pn) {
      const float* pp = pos + ((size_t)b * Pn + p) * 3;
      float dx = pp[0] - cx, dy = pp[1] - cy, dz = pp[2] - cz;
      float dd = dx * dx + dy * dy + dz * dz;
      d = (dd <= r2) ? dd : BIGF;
    }
    d2s[p] = d;
  }
  __syncthreads();
  for (int k = 0; k < KN; ++k) {
    float bv = BIGF; int bi = 0x7fffffff;
    for (int p = t; p < 512; p += 128) {
      float v = d2s[p];
      if (v < bv || (v == bv && p < bi)) { bv = v; bi = p; }
    }
    rv[t] = bv; ri[t] = bi;
    __syncthreads();
    for (int off = 64; off > 0; off >>= 1) {
      if (t < off) {
        float ov = rv[t + off]; int oi = ri[t + off];
        if (ov < rv[t] || (ov == rv[t] && oi < ri[t])) { rv[t] = ov; ri[t] = oi; }
      }
      __syncthreads();
    }
    if (t == 0) {
      if (rv[0] >= 1.0e37f) {
        nbr[(size_t)cidx * KN + k] = -1;
      } else {
        nbr[(size_t)cidx * KN + k] = ri[0];
        d2s[ri[0]] = BIGF;
      }
    }
    __syncthreads();
  }
}

// ---------------------------------------------------------------- WMMA fragment loaders (ISA 7.12.2)
// A 16x32 f16 tile from row-major LDS: per lane the fragment is two contiguous
// 16B runs (halfs p[0..7] and p[16..23]) -> 2x ds_load_b128.
__device__ __forceinline__ v16h load_a_lds(const half_t* buf, int ldk, int rowBase, int kBase) {
  int lane = threadIdx.x & 31;
  int grp  = lane >> 4;
  int r    = lane & 15;
  const half_t* p = buf + (rowBase + r) * ldk + kBase + grp * 8;
  v8h lo = *(const v8h*)(p);        // K = grp*8 + 0..7   (VGPR 0..3)
  v8h hi = *(const v8h*)(p + 16);   // K = grp*8 + 16..23 (VGPR 4..7)
  return __builtin_shufflevector(lo, hi, 0, 1, 2, 3, 4, 5, 6, 7,
                                 8, 9, 10, 11, 12, 13, 14, 15);
}

// B 32x16 f16 fragment from pre-swizzled weights: one 32B contiguous chunk per lane.
__device__ __forceinline__ v16h load_b_packed(const half_t* __restrict__ Wpk, int frag) {
  int lane = threadIdx.x & 31;
  const half_t* p = Wpk + ((size_t)frag * 32 + lane) * 16;
  v8h lo = *(const v8h*)(p);
  v8h hi = *(const v8h*)(p + 8);
  return __builtin_shufflevector(lo, hi, 0, 1, 2, 3, 4, 5, 6, 7,
                                 8, 9, 10, 11, 12, 13, 14, 15);
}

// ---------------------------------------------------------------- fused GEMM + bias + ReLU + affine
// A: [64 x Kd] f16 in LDS (row-major, stride Kd). Wpk: packed fragments (ct-major, ks inner).
// out = relu(A@W + b) * (g*rsqrt(1+eps)) + bt, written f16 to LDS (stride Cout).
__device__ __forceinline__ void gemm_relu_affine_lds(
    const half_t* aBuf, int Kd, const half_t* __restrict__ Wpk, int Cout,
    const float* __restrict__ bias, const float* __restrict__ gam,
    const float* __restrict__ bet, half_t* outBuf) {
  int wave = threadIdx.x >> 5;
  int lane = threadIdx.x & 31;
  int grp  = lane >> 4;
  int ln   = lane & 15;
  const float rs = rsqrtf(1.0f + EPSF);
  int nKs    = Kd >> 5;
  int nTiles = 4 * (Cout >> 4);          // multiple of 8 -> uniform trip count per wave
  for (int t = wave; t < nTiles; t += 8) {
    int rowBase = (t & 3) << 4;
    int colTile = t >> 2;
    v8f acc = {0.f, 0.f, 0.f, 0.f, 0.f, 0.f, 0.f, 0.f};
    int fragBase = colTile * nKs;
    for (int ks = 0; ks < nKs; ++ks) {
      v16h a = load_a_lds(aBuf, Kd, rowBase, ks << 5);
      v16h b = load_b_packed(Wpk, fragBase + ks);
      acc = __builtin_amdgcn_wmma_f32_16x16x32_f16(false, a, false, b, (short)0, acc,
                                                   false, false);
    }
    int col = (colTile << 4) + ln;
    float bv = bias[col];
    float sv = gam[col] * rs;
    float tv = bet[col];
#pragma unroll
    for (int v = 0; v < 8; ++v) {
      int row = rowBase + grp * 8 + v;
      float z = acc[v] + bv;
      z = fmaxf(z, 0.0f);
      z = z * sv + tv;
      outBuf[row * Cout + col] = (half_t)z;
    }
  }
}

// Same, but epilogue writes f32 to global (final 512x1024 layer).
__device__ __forceinline__ void gemm_relu_affine_global(
    const half_t* aBuf, int Kd, const half_t* __restrict__ Wpk, int Cout,
    const float* __restrict__ bias, const float* __restrict__ gam,
    const float* __restrict__ bet, float* __restrict__ gout) {
  int wave = threadIdx.x >> 5;
  int lane = threadIdx.x & 31;
  int grp  = lane >> 4;
  int ln   = lane & 15;
  const float rs = rsqrtf(1.0f + EPSF);
  int nKs    = Kd >> 5;
  int nTiles = 4 * (Cout >> 4);
  for (int t = wave; t < nTiles; t += 8) {
    int rowBase = (t & 3) << 4;
    int colTile = t >> 2;
    v8f acc = {0.f, 0.f, 0.f, 0.f, 0.f, 0.f, 0.f, 0.f};
    int fragBase = colTile * nKs;
    for (int ks = 0; ks < nKs; ++ks) {
      v16h a = load_a_lds(aBuf, Kd, rowBase, ks << 5);
      v16h b = load_b_packed(Wpk, fragBase + ks);
      acc = __builtin_amdgcn_wmma_f32_16x16x32_f16(false, a, false, b, (short)0, acc,
                                                   false, false);
    }
    int col = (colTile << 4) + ln;
    float bv = bias[col];
    float sv = gam[col] * rs;
    float tv = bet[col];
#pragma unroll
    for (int v = 0; v < 8; ++v) {
      int row = rowBase + grp * 8 + v;
      float z = acc[v] + bv;
      z = fmaxf(z, 0.0f);
      z = z * sv + tv;
      gout[(size_t)row * Cout + col] = z;
    }
  }
}

// ---------------------------------------------------------------- SA1: gather + MLP[32->64->64->128] + maxpool
__global__ void __launch_bounds__(256) sa1_kernel(
    const float* __restrict__ pos, const float* __restrict__ centers,
    const int* __restrict__ nbr,
    const half_t* __restrict__ w0, const float* b0, const float* g0, const float* t0,
    const half_t* __restrict__ w1, const float* b1, const float* g1, const float* t1,
    const half_t* __restrict__ w2, const float* b2, const float* g2, const float* t2,
    half_t* __restrict__ x1out) {
  __shared__ __align__(16) half_t buf0[64 * 64];
  __shared__ __align__(16) half_t buf1[64 * 128];
  __shared__ int valid[64];
  int cidx = blockIdx.x;           // b*S1 + s
  int b = cidx / S1;
  int tid = threadIdx.x;
  for (int i = tid; i < 64 * 32; i += 256) buf0[i] = (half_t)0.0f;
  float cx = centers[(size_t)cidx * 3 + 0];
  float cy = centers[(size_t)cidx * 3 + 1];
  float cz = centers[(size_t)cidx * 3 + 2];
  __syncthreads();
  if (tid < 64) {
    int n = nbr[(size_t)cidx * KN + tid];
    valid[tid] = (n >= 0) ? 1 : 0;
    if (n >= 0) {
      const float* pp = pos + ((size_t)b * P1 + n) * 3;
      half_t* row = buf0 + tid * 32;
      row[0] = (half_t)pp[0]; row[1] = (half_t)pp[1]; row[2] = (half_t)pp[2];
      row[3] = (half_t)(pp[0] - cx); row[4] = (half_t)(pp[1] - cy); row[5] = (half_t)(pp[2] - cz);
    }
  }
  __syncthreads();
  gemm_relu_affine_lds(buf0, 32, w0, 64, b0, g0, t0, buf1);
  __syncthreads();
  gemm_relu_affine_lds(buf1, 64, w1, 64, b1, g1, t1, buf0);
  __syncthreads();
  gemm_relu_affine_lds(buf0, 64, w2, 128, b2, g2, t2, buf1);
  __syncthreads();
  for (int c = tid; c < 128; c += 256) {
    float m = -BIGF; bool found = false;
    for (int r = 0; r < 64; ++r)
      if (valid[r]) { m = fmaxf(m, (float)buf1[r * 128 + c]); found = true; }
    x1out[(size_t)cidx * 128 + c] = (half_t)(found ? m : 0.0f);
  }
}

// ---------------------------------------------------------------- SA2: gather + MLP[160->128->128->256] + maxpool
__global__ void __launch_bounds__(256) sa2_kernel(
    const half_t* __restrict__ x1, const float* __restrict__ pos1,
    const float* __restrict__ centers,
    const int* __restrict__ nbr,
    const half_t* __restrict__ w0, const float* b0, const float* g0, const float* t0,
    const half_t* __restrict__ w1, const float* b1, const float* g1, const float* t1,
    const half_t* __restrict__ w2, const float* b2, const float* g2, const float* t2,
    half_t* __restrict__ x2out) {
  __shared__ __align__(16) half_t buf0[64 * 160];
  __shared__ __align__(16) half_t buf1[64 * 256];
  __shared__ int valid[64];
  int cidx = blockIdx.x;           // b*S2 + s
  int b = cidx / S2;
  int tid = threadIdx.x;
  for (int i = tid; i < 64 * 160; i += 256) buf0[i] = (half_t)0.0f;
  float cx = centers[(size_t)cidx * 3 + 0];
  float cy = centers[(size_t)cidx * 3 + 1];
  float cz = centers[(size_t)cidx * 3 + 2];
  __syncthreads();
  for (int i = tid; i < 64 * 128; i += 256) {     // xj gather (128 f16 feats per row)
    int r = i >> 7, c = i & 127;
    int n = nbr[(size_t)cidx * KN + r];
    if (n >= 0) buf0[r * 160 + c] = x1[((size_t)b * S1 + n) * 128 + c];
  }
  if (tid < 64) {                                  // pj + valid flag
    int n = nbr[(size_t)cidx * KN + tid];
    valid[tid] = (n >= 0) ? 1 : 0;
    if (n >= 0) {
      const float* pp = pos1 + ((size_t)b * S1 + n) * 3;
      half_t* row = buf0 + tid * 160;
      row[128] = (half_t)(pp[0] - cx);
      row[129] = (half_t)(pp[1] - cy);
      row[130] = (half_t)(pp[2] - cz);
    }
  }
  __syncthreads();
  gemm_relu_affine_lds(buf0, 160, w0, 128, b0, g0, t0, buf1);
  __syncthreads();
  gemm_relu_affine_lds(buf1, 128, w1, 128, b1, g1, t1, buf0);
  __syncthreads();
  gemm_relu_affine_lds(buf0, 128, w2, 256, b2, g2, t2, buf1);
  __syncthreads();
  for (int c = tid; c < 256; c += 256) {
    float m = -BIGF; bool found = false;
    for (int r = 0; r < 64; ++r)
      if (valid[r]) { m = fmaxf(m, (float)buf1[r * 256 + c]); found = true; }
    x2out[(size_t)cidx * 256 + c] = (half_t)(found ? m : 0.0f);
  }
}

// ---------------------------------------------------------------- MLP3 [288->256->512->1024] on 64-row blocks
__global__ void __launch_bounds__(256) mlp3_kernel(
    const half_t* __restrict__ x2, const float* __restrict__ centers2,
    const half_t* __restrict__ w0, const float* b0, const float* g0, const float* t0,
    const half_t* __restrict__ w1, const float* b1, const float* g1, const float* t1,
    const half_t* __restrict__ w2, const float* b2, const float* g2, const float* t2,
    float* __restrict__ h) {
  extern __shared__ __align__(16) half_t dbuf[];
  half_t* buf0 = dbuf;             // 64 * 512
  half_t* buf1 = dbuf + 64 * 512;  // 64 * 512
  int rowBase = blockIdx.x * 64;   // 26 blocks of 64 rows (rows = b*S2 + s)
  int tid = threadIdx.x;
  for (int i = tid; i < 64 * 288; i += 256) buf0[i] = (half_t)0.0f;
  __syncthreads();
  for (int i = tid; i < 64 * 256; i += 256) {
    int r = i >> 8, c = i & 255;
    buf0[r * 288 + c] = x2[(size_t)(rowBase + r) * 256 + c];
  }
  if (tid < 64) {
    int ri = rowBase + tid;
    buf0[tid * 288 + 256] = (half_t)centers2[(size_t)ri * 3 + 0];
    buf0[tid * 288 + 257] = (half_t)centers2[(size_t)ri * 3 + 1];
    buf0[tid * 288 + 258] = (half_t)centers2[(size_t)ri * 3 + 2];
  }
  __syncthreads();
  gemm_relu_affine_lds(buf0, 288, w0, 256, b0, g0, t0, buf1);
  __syncthreads();
  gemm_relu_affine_lds(buf1, 256, w1, 512, b1, g1, t1, buf0);
  __syncthreads();
  gemm_relu_affine_global(buf0, 512, w2, 1024, b2, g2, t2, h + (size_t)rowBase * 1024);
}

// ---------------------------------------------------------------- final max over the 26 centers
__global__ void final_pool_kernel(const float* __restrict__ h, float* __restrict__ out) {
  int b = blockIdx.x;
  for (int c = threadIdx.x; c < 1024; c += 256) {
    float m = -BIGF;
    for (int s = 0; s < S2; ++s)
      m = fmaxf(m, h[((size_t)b * S2 + s) * 1024 + c]);
    out[(size_t)b * 1024 + c] = m;
  }
}

// ---------------------------------------------------------------- host
extern "C" void kernel_launch(void* const* d_in, const int* in_sizes, int n_in,
                              void* d_out, int out_size, void* d_ws, size_t ws_size,
                              hipStream_t stream) {
  (void)in_sizes; (void)n_in; (void)out_size; (void)ws_size;
  const float* x = (const float*)d_in[0];
  const float* Wp[9]; const float* Bp[9]; const float* Gp[9]; const float* Tp[9];
  for (int m = 0; m < 3; ++m)
    for (int l = 0; l < 3; ++l) {
      int li = m * 3 + l;
      int base = 1 + li * 4;
      Wp[li] = (const float*)d_in[base + 0];
      Bp[li] = (const float*)d_in[base + 1];
      Gp[li] = (const float*)d_in[base + 2];
      Tp[li] = (const float*)d_in[base + 3];
    }
  static const int kin[9]  = {6, 64, 64, 131, 128, 128, 259, 256, 512};
  static const int kpad[9] = {32, 64, 64, 160, 128, 128, 288, 256, 512};
  static const int cout[9] = {64, 64, 128, 128, 128, 256, 256, 512, 1024};

  char* ws = (char*)d_ws;
  size_t off = 0;
  auto alloc = [&](size_t bytes) -> void* {
    void* p = ws + off;
    off += (bytes + 255) & ~(size_t)255;
    return p;
  };
  half_t* Wh[9];
  for (int i = 0; i < 9; ++i)
    Wh[i] = (half_t*)alloc((size_t)kpad[i] * cout[i] * sizeof(half_t));
  float* centers1 = (float*)alloc((size_t)BN * S1 * 3 * sizeof(float));
  float* centers2 = (float*)alloc((size_t)BN * S2 * 3 * sizeof(float));
  int*   nbr1     = (int*)alloc((size_t)BN * S1 * KN * sizeof(int));
  int*   nbr2     = (int*)alloc((size_t)BN * S2 * KN * sizeof(int));
  half_t* x1      = (half_t*)alloc((size_t)BN * S1 * 128 * sizeof(half_t));
  half_t* x2      = (half_t*)alloc((size_t)BN * S2 * 256 * sizeof(half_t));
  float* h        = (float*)alloc((size_t)BN * S2 * 1024 * sizeof(float));

  for (int i = 0; i < 9; ++i) {
    int total = kpad[i] * cout[i];
    pack_w_kernel<<<(total + 255) / 256, 256, 0, stream>>>(Wp[i], Wh[i], kin[i], cout[i],
                                                           kpad[i] >> 5);
  }
  fps_kernel<<<BN, 512, 0, stream>>>(x, centers1, P1, S1);
  knn_kernel<<<BN * S1, 128, 0, stream>>>(x, centers1, nbr1, P1, S1, 0.2f * 0.2f);
  sa1_kernel<<<BN * S1, 256, 0, stream>>>(x, centers1, nbr1,
      Wh[0], Bp[0], Gp[0], Tp[0],
      Wh[1], Bp[1], Gp[1], Tp[1],
      Wh[2], Bp[2], Gp[2], Tp[2], x1);
  fps_kernel<<<BN, 512, 0, stream>>>(centers1, centers2, S1, S2);
  knn_kernel<<<BN * S2, 128, 0, stream>>>(centers1, centers2, nbr2, S1, S2, 0.4f * 0.4f);
  sa2_kernel<<<BN * S2, 256, 0, stream>>>(x1, centers1, centers2, nbr2,
      Wh[3], Bp[3], Gp[3], Tp[3],
      Wh[4], Bp[4], Gp[4], Tp[4],
      Wh[5], Bp[5], Gp[5], Tp[5], x2);
  size_t mlp3_lds = (size_t)2 * 64 * 512 * sizeof(half_t);  // 128 KB dynamic LDS
  mlp3_kernel<<<(BN * S2) / 64, 256, mlp3_lds, stream>>>(x2, centers2,
      Wh[6], Bp[6], Gp[6], Tp[6],
      Wh[7], Bp[7], Gp[7], Tp[7],
      Wh[8], Bp[8], Gp[8], Tp[8], h);
  final_pool_kernel<<<BN, 256, 0, stream>>>(h, (float*)d_out);
}
